// AntisymmetricRNN_70549132804852
// MI455X (gfx1250) — matrix operand order repaired and checked by reading, not compiled, not run on hardware
//
#include <hip/hip_runtime.h>
#include <hip/hip_bf16.h>
#include <math.h>

// Problem constants (from reference)
#define B_    128
#define T_    2048
#define DIN   256
#define NU    512
#define NOUT  64
#define EPSF   0.01f
#define GAMMAF 0.01f

typedef __attribute__((ext_vector_type(16))) __bf16 v16bf;
typedef __attribute__((ext_vector_type(8)))  __bf16 v8bf;
typedef __attribute__((ext_vector_type(8)))  float  v8f;
typedef __attribute__((ext_vector_type(4)))  float  v4f;

// ---- workspace layout (bytes) ----
static const size_t OFF_A  = 0;                         // A  B-layout bf16 512KB
static const size_t OFF_VH = 512*512*2;                 // Vh B-layout bf16 256KB
static const size_t OFF_VZ = OFF_VH + 256*512*2;        // Vz B-layout bf16 256KB
static const size_t OFF_PH = OFF_VZ + 256*512*2;        // proj vh bf16 256MB
static const size_t OFF_PZ = OFF_PH + (size_t)T_*B_*NU*2; // proj vz bf16 256MB

union V16U { v16bf v; v8bf h[2]; };

__device__ __forceinline__ v8f wmma_bf16(v16bf a, v16bf b, v8f c) {
  return __builtin_amdgcn_wmma_f32_16x16x32_bf16(false, a, false, b, (short)0, c,
                                                 false, false);
}

// WMMA A-operand gather from row-major [16 x stride] bf16 (contiguous k order, same as the B packing):
// lanes 0-15: M=lane, K=kb+{0..15}; lanes 16-31: M=lane-16, K=kb+{16..31}.
__device__ __forceinline__ v16bf load_a_tile(const __bf16* base, int stride,
                                             int kb, int lane) {
  int m   = lane & 15;
  int off = (lane >> 4) ? 16 : 0;
  const __bf16* p = base + m * stride + kb + off;
  V16U u;
  u.h[0] = *(const v8bf*)(p);
  u.h[1] = *(const v8bf*)(p + 8);
  return u.v;
}

// ---------------------------------------------------------------------------
// Kernel 1: A = W - W^T - gamma*I pre-swizzled into WMMA B-operand layout.
// Chunk c = ntile*16 + kblock: 32 lanes x 16 bf16; lane<16: N=ntile*16+lane,
// K=kblock*32+{0..15}; lane>=16: N=ntile*16+lane-16, K=kblock*32+{16..31}.
// ---------------------------------------------------------------------------
__global__ __launch_bounds__(256) void build_A_kernel(const float* __restrict__ W,
                                                      __bf16* __restrict__ wsA) {
  int tid   = blockIdx.x * blockDim.x + threadIdx.x;   // 512 chunks * 32 lanes
  int c     = tid >> 5, lane = tid & 31;
  int ntile = c >> 4,  kblock = c & 15;
  int n     = ntile * 16 + (lane & 15);
  int kbase = kblock * 32 + ((lane >> 4) ? 16 : 0);
  V16U u;
#pragma unroll
  for (int j = 0; j < 16; ++j) {
    int k = kbase + j;
    float v = W[k * NU + n] - W[n * NU + k] - ((k == n) ? GAMMAF : 0.0f);
    u.v[j] = (__bf16)v;
  }
  *(v16bf*)(wsA + ((size_t)c * 512 + lane * 16)) = u.v;
}

// ---------------------------------------------------------------------------
// Kernel 2: Vh_w^T / Vz_w^T into B-operand layout bf16.
// ---------------------------------------------------------------------------
__global__ __launch_bounds__(256) void build_V_kernel(const float* __restrict__ Vh_w,
                                                      const float* __restrict__ Vz_w,
                                                      __bf16* __restrict__ wsVh,
                                                      __bf16* __restrict__ wsVz) {
  int tid  = blockIdx.x * blockDim.x + threadIdx.x;    // 2*256 chunks * 32 lanes
  int c2   = tid >> 5, lane = tid & 31;
  const float* V   = (c2 < 256) ? Vh_w : Vz_w;
  __bf16*      dst = (c2 < 256) ? wsVh : wsVz;
  int c     = c2 & 255;
  int ntile = c >> 3, kblock = c & 7;
  int n     = ntile * 16 + (lane & 15);
  int kbase = kblock * 32 + ((lane >> 4) ? 16 : 0);
  V16U u;
#pragma unroll
  for (int j = 0; j < 16; ++j)
    u.v[j] = (__bf16)V[n * DIN + kbase + j];           // B[k][n] = V_w[n][k]
  *(v16bf*)(dst + ((size_t)c * 512 + lane * 16)) = u.v;
}

// ---------------------------------------------------------------------------
// Kernel 3: projection GEMMs.  512-thread blocks; __launch_bounds__(512) lifts
// the VGPR cap to 256 so the 128-VGPR weight slices stay resident (no spills).
// Each block: one 16-row batch tile, 8 timesteps, HALF the ntiles.
// Grid = (T/8)*8*2.
// ---------------------------------------------------------------------------
__global__ __launch_bounds__(512) void proj_kernel(const float* __restrict__ x,
                                                   const __bf16* __restrict__ wsVh,
                                                   const __bf16* __restrict__ wsVz,
                                                   const float* __restrict__ Vh_b,
                                                   const float* __restrict__ Vz_b,
                                                   __bf16* __restrict__ ph,
                                                   __bf16* __restrict__ pz) {
  __shared__ __bf16 xs[16 * DIN];                      // 8 KB staged x tile
  int nh    = blockIdx.x & 1;                          // ntile half 0/1
  int bt    = (blockIdx.x >> 1) & 7;                   // batch tile 0..7
  int tg    = blockIdx.x >> 4;                         // time group 0..255
  int wave  = threadIdx.x >> 5, lane = threadIdx.x & 31;
  int ntile = nh * 16 + wave;
  int nlane = ntile * 16 + (lane & 15);
  float bh = Vh_b[nlane], bz = Vz_b[nlane];

  v16bf wh[8], wz[8];                                  // 128 VGPRs, resident
#pragma unroll
  for (int kk = 0; kk < 8; ++kk) {
    size_t c = (size_t)ntile * 8 + kk;
    wh[kk] = *(const v16bf*)(wsVh + c * 512 + lane * 16);
    wz[kk] = *(const v16bf*)(wsVz + c * 512 + lane * 16);
  }

  for (int tt = 0; tt < 8; ++tt) {
    int t = tg * 8 + tt;
    __syncthreads();                                   // xs free to overwrite
    {
      int i = threadIdx.x * 8;                         // 8 f32 per thread
      int m = i >> 8, d = i & 255;
      int b = bt * 16 + m;
      const float* xp = x + ((size_t)b * T_ + t) * DIN + d;
      v4f v0 = *(const v4f*)(xp);
      v4f v1 = *(const v4f*)(xp + 4);
      xs[m * DIN + d + 0] = (__bf16)v0.x;
      xs[m * DIN + d + 1] = (__bf16)v0.y;
      xs[m * DIN + d + 2] = (__bf16)v0.z;
      xs[m * DIN + d + 3] = (__bf16)v0.w;
      xs[m * DIN + d + 4] = (__bf16)v1.x;
      xs[m * DIN + d + 5] = (__bf16)v1.y;
      xs[m * DIN + d + 6] = (__bf16)v1.z;
      xs[m * DIN + d + 7] = (__bf16)v1.w;
    }
    __syncthreads();

    v8f acch = {}, accz = {};
#pragma unroll
    for (int kk = 0; kk < 8; ++kk) {
      v16bf a = load_a_tile(xs, DIN, kk * 32, lane);
      acch = wmma_bf16(a, wh[kk], acch);
      accz = wmma_bf16(a, wz[kk], accz);
    }
    v8bf oh, oz;
#pragma unroll
    for (int r = 0; r < 8; ++r) {
      oh[r] = (__bf16)(acch[r] + bh);
      oz[r] = (__bf16)(accz[r] + bz);
    }
    size_t chunk = (((size_t)t * 8 + bt) * 32 + ntile);
    *(v8bf*)(ph + chunk * 256 + lane * 8) = oh;
    *(v8bf*)(pz + chunk * 256 + lane * 8) = oz;
  }
}

// ---------------------------------------------------------------------------
// Kernel 4: sequential scan.  8 blocks x 1024 threads = 32 waves; wave i owns
// N-tile i.  A-slice storage split to fit the WGP exactly: kk=0..7 (64 VGPRs/
// wave) in registers, kk=8..15 in 256 KB of LDS.  h: f32 in registers
// (C-layout) + bf16 mirror in LDS as WMMA A-operand.  f32 epilogue buffer
// overlaps the dead A region of LDS after the loop.
// ---------------------------------------------------------------------------
#define A_LDS_BYTES (32 * 8 * 512 * 2)                 // 256 KB
#define HBF_OFF     A_LDS_BYTES                        // 16 KB after A

__global__ __launch_bounds__(1024) void rnn_scan_kernel(const __bf16* __restrict__ wsA,
                                                        const __bf16* __restrict__ ph,
                                                        const __bf16* __restrict__ pz,
                                                        const float* __restrict__ fc_w,
                                                        const float* __restrict__ fc_b,
                                                        float* __restrict__ out) {
  __shared__ __attribute__((aligned(32))) char smem[A_LDS_BYTES + 16 * NU * 2]; // 272 KB
  __bf16* Alds = (__bf16*)smem;                        // [ntile][kk-8][lane*16]
  __bf16* hbf  = (__bf16*)(smem + HBF_OFF);            // 16x512 bf16
  float*  hf   = (float*)smem;                         // epilogue, overlaps Alds

  int bt    = blockIdx.x;
  int wave  = threadIdx.x >> 5, lane = threadIdx.x & 31;
  int ntile = wave;
  int nlane = ntile * 16 + (lane & 15);
  int moff  = (lane >> 4) ? 8 : 0;                     // C-layout row offset

  v16bf Areg[8];                                       // kk = 0..7 resident
#pragma unroll
  for (int kk = 0; kk < 8; ++kk)
    Areg[kk] = *(const v16bf*)(wsA + ((size_t)(ntile * 16 + kk) * 512 + lane * 16));
#pragma unroll
  for (int kk = 8; kk < 16; ++kk) {                    // kk = 8..15 -> LDS
    v16bf tA = *(const v16bf*)(wsA + ((size_t)(ntile * 16 + kk) * 512 + lane * 16));
    *(v16bf*)(Alds + ((size_t)(ntile * 8 + (kk - 8)) * 512 + lane * 16)) = tA;
  }

  for (int i = threadIdx.x; i < 16 * NU; i += 1024) hbf[i] = (__bf16)0.0f;
  float hreg[8];
#pragma unroll
  for (int r = 0; r < 8; ++r) hreg[r] = 0.0f;
  __syncthreads();

  for (int t = 0; t < T_; ++t) {
    v8f acc = {};
#pragma unroll
    for (int kk = 0; kk < 8; ++kk) {                   // VGPR-resident B half
      v16bf a = load_a_tile(hbf, NU, kk * 32, lane);
      acc = wmma_bf16(a, Areg[kk], acc);
    }
#pragma unroll
    for (int kk = 8; kk < 16; ++kk) {                  // LDS-resident B half
      v16bf a = load_a_tile(hbf, NU, kk * 32, lane);
      v16bf b = *(const v16bf*)(Alds + ((size_t)(ntile * 8 + (kk - 8)) * 512 + lane * 16));
      acc = wmma_bf16(a, b, acc);
    }
    size_t chunk = (((size_t)t * 8 + bt) * 32 + ntile);
    v8bf vh = *(const v8bf*)(ph + chunk * 256 + lane * 8);
    v8bf vz = *(const v8bf*)(pz + chunk * 256 + lane * 8);
    __syncthreads();                                   // hbf reads complete
#pragma unroll
    for (int r = 0; r < 8; ++r) {
      float u1 = acc[r] + (float)vh[r];
      float u2 = acc[r] + (float)vz[r];
      float f  = tanhf(u1) * (1.0f / (1.0f + expf(-u2)));
      hreg[r] += EPSF * f;
      hbf[(moff + r) * NU + nlane] = (__bf16)hreg[r];
    }
    __syncthreads();                                   // hbf writes visible
  }

  // epilogue: out = h @ fc_w^T + fc_b (tiny, f32 exact); hf overlaps dead Alds
#pragma unroll
  for (int r = 0; r < 8; ++r) hf[(moff + r) * NU + nlane] = hreg[r];
  __syncthreads();
  int m = threadIdx.x >> 6, o = threadIdx.x & 63;      // 16 x 64 outputs
  float s = fc_b[o];
  for (int n2 = 0; n2 < NU; ++n2)
    s = fmaf(hf[m * NU + n2], fc_w[o * NU + n2], s);
  out[(bt * 16 + m) * NOUT + o] = s;
}

// ---------------------------------------------------------------------------
extern "C" void kernel_launch(void* const* d_in, const int* in_sizes, int n_in,
                              void* d_out, int out_size, void* d_ws, size_t ws_size,
                              hipStream_t stream) {
  const float* x    = (const float*)d_in[0];
  const float* Vh_w = (const float*)d_in[1];
  const float* Vh_b = (const float*)d_in[2];
  const float* Vz_w = (const float*)d_in[3];
  const float* Vz_b = (const float*)d_in[4];
  const float* W    = (const float*)d_in[5];
  const float* fc_w = (const float*)d_in[6];
  const float* fc_b = (const float*)d_in[7];
  float* out = (float*)d_out;

  char* ws = (char*)d_ws;
  __bf16* wsA  = (__bf16*)(ws + OFF_A);
  __bf16* wsVh = (__bf16*)(ws + OFF_VH);
  __bf16* wsVz = (__bf16*)(ws + OFF_VZ);
  __bf16* pvh  = (__bf16*)(ws + OFF_PH);
  __bf16* pvz  = (__bf16*)(ws + OFF_PZ);
  // requires ws_size >= OFF_PZ + T*B*NU*2 (~538 MB)

  build_A_kernel<<<64, 256, 0, stream>>>(W, wsA);
  build_V_kernel<<<64, 256, 0, stream>>>(Vh_w, Vz_w, wsVh, wsVz);
  proj_kernel<<<(T_ / 8) * 8 * 2, 512, 0, stream>>>(x, wsVh, wsVz, Vh_b, Vz_b, pvh, pvz);
  rnn_scan_kernel<<<8, 1024, 0, stream>>>(wsA, pvh, pvz, fc_w, fc_b, out);
}
